// WindowAttention_17540646437103
// MI455X (gfx1250) — matrix operand order repaired and checked by reading, not compiled, not run on hardware
//
#include <hip/hip_runtime.h>
#include <hip/hip_bf16.h>

typedef __attribute__((ext_vector_type(16))) __bf16 bf16x16;
typedef __attribute__((ext_vector_type(8)))  float  f32x8;
typedef __attribute__((ext_vector_type(4)))  unsigned int u32x4;
typedef __attribute__((ext_vector_type(8)))  int    i32x8;
typedef __attribute__((ext_vector_type(4)))  int    i32x4;

#define SCALE_Q 0.17677669529663687f  // 32^-0.5
#define M_TOT   100352                 // 2048*49
#define TS      40                     // LDS row stride (bf16 elems), 80B = 16B-aligned chunks

__device__ __forceinline__ unsigned short f2bf(float f) {
    union { float f; unsigned u; } x{f};
    unsigned r = x.u + 0x7FFFu + ((x.u >> 16) & 1u);   // round-to-nearest-even
    return (unsigned short)(r >> 16);
}

__device__ __forceinline__ f32x8 wmma_bf16(bf16x16 a, bf16x16 b, f32x8 c) {
    return __builtin_amdgcn_wmma_f32_16x16x32_bf16(false, a, false, b, (short)0, c, false, false);
}

// A fragment (16x32 bf16): rows [row0,row0+16), K cols [k0,k0+32).
__device__ __forceinline__ bf16x16 load_a_frag(const unsigned short* lds, int row0, int k0,
                                               int stride, int lane) {
    int half = lane >> 4;
    const unsigned short* p = lds + (row0 + (lane & 15)) * stride + k0;
    union { bf16x16 v; uint4 q[2]; } u;
    u.q[0] = *reinterpret_cast<const uint4*>(p + 16 * half);
    u.q[1] = *reinterpret_cast<const uint4*>(p + 16 * half + 8);
    return u.v;
}

// B fragment (32x16) where B = src^T, src row-major [n][k].
__device__ __forceinline__ bf16x16 load_b_frag_rows(const unsigned short* lds, int n0, int k0,
                                                    int stride, int lane) {
    int half = lane >> 4;
    const unsigned short* p = lds + (n0 + (lane & 15)) * stride + k0 + 16 * half;
    union { bf16x16 v; uint4 q[2]; } u;
    u.q[0] = *reinterpret_cast<const uint4*>(p);
    u.q[1] = *reinterpret_cast<const uint4*>(p + 8);
    return u.v;
}

// B fragment (32x16) where B = src directly, src row-major [k][n] (v matrix).
__device__ __forceinline__ bf16x16 load_b_frag_cols(const unsigned short* lds, int n0, int k0,
                                                    int stride, int lane) {
    int half = lane >> 4;
    int n  = n0 + (lane & 15);
    int jb = k0 + 16 * half;
    union { bf16x16 v; unsigned short s[16]; } u;
#pragma unroll
    for (int t = 0; t < 16; ++t) u.s[t] = lds[(jb + t) * stride + n];
    return u.v;
}

// Tensor Data Mover: 2D tile 32 x 49 (bf16), global row stride 1152 elems,
// LDS pad: 16 DWORDs (one 64B row) -> +4 DWORDs (16B) => 80B LDS row stride == TS.
__device__ __forceinline__ void tdm_load_qkv_tile(unsigned lds_off, const unsigned short* gptr) {
    unsigned long long ga = (unsigned long long)gptr;
    u32x4 g0;
    g0[0] = 1u;                                             // count=1
    g0[1] = lds_off;                                        // lds_addr (bytes)
    g0[2] = (unsigned)(ga & 0xFFFFFFFFu);                   // global_addr[31:0]
    g0[3] = (unsigned)((ga >> 32) & 0x01FFFFFFu) | (2u << 30); // global_addr[56:32], type=2
    i32x8 g1;
    g1[0] = (1 << 16)        // data_size = 2 bytes
          | (1 << 20)        // pad_enable
          | (3 << 22)        // pad_interval: 16 DWORDs
          | (3 << 25);       // pad_amount: 4 DWORDs
    g1[1] = (int)(32u << 16);    // tensor_dim0 = 32
    g1[2] = (int)(49u << 16);    // tensor_dim1 = 49
    g1[3] = (int)(32u << 16);    // tile_dim0 = 32
    g1[4] = 49;                  // tile_dim1 = 49, tile_dim2 = 0
    g1[5] = 1152;                // tensor_dim0_stride = 1152 elems
    g1[6] = 0;
    g1[7] = 0;
    i32x4 z4 = {0, 0, 0, 0};
#if __clang_major__ >= 23
    i32x8 z8 = {0, 0, 0, 0, 0, 0, 0, 0};
    __builtin_amdgcn_tensor_load_to_lds(g0, g1, z4, z4, z8, 0);
#else
    __builtin_amdgcn_tensor_load_to_lds(g0, g1, z4, z4, 0);
#endif
}

// ---------------- Kernel 0: one-shot fp32 -> bf16 weight conversion
__global__ __launch_bounds__(256) void convert_weights(const float* __restrict__ qkv_w,
                                                       const float* __restrict__ proj_w,
                                                       unsigned short* __restrict__ wqkv,
                                                       unsigned short* __restrict__ wproj) {
    int i = (blockIdx.x * 256 + threadIdx.x) * 4;
    if (i < 442368) {   // 1152*384
        float4 v = *reinterpret_cast<const float4*>(qkv_w + i);
        wqkv[i + 0] = f2bf(v.x); wqkv[i + 1] = f2bf(v.y);
        wqkv[i + 2] = f2bf(v.z); wqkv[i + 3] = f2bf(v.w);
    } else {
        int j = i - 442368;
        if (j < 147456) {   // 384*384
            float4 v = *reinterpret_cast<const float4*>(proj_w + j);
            wproj[j + 0] = f2bf(v.x); wproj[j + 1] = f2bf(v.y);
            wproj[j + 2] = f2bf(v.z); wproj[j + 3] = f2bf(v.w);
        }
    }
}

// ---------------- Kernel 1: qkv = (x*stagein) @ qkv_w^T + qkv_b ; q *= SCALE ; bf16 out
__global__ __launch_bounds__(128) void qkv_kernel(const float* __restrict__ x,
                                                  const float* __restrict__ stagein,
                                                  const unsigned short* __restrict__ wqkv,
                                                  const float* __restrict__ qkv_b,
                                                  unsigned short* __restrict__ qkv_out) {
    __shared__ unsigned short lA[64 * TS];
    __shared__ unsigned short lB[64 * TS];
    const int tid = threadIdx.x, lane = tid & 31, wave = tid >> 5;
    const int m0 = blockIdx.y * 64;
    const int n0 = blockIdx.x * 64;
    f32x8 acc[4] = {};

    for (int kk = 0; kk < 384; kk += 32) {
        const int r = tid >> 1, hf = tid & 1;
        {   // stage A = x*stagein (fp32 -> bf16), vectorized float4
            const float* sx = x       + (size_t)(m0 + r) * 384 + kk + hf * 16;
            const float* sg = stagein + (size_t)(m0 + r) * 384 + kk + hf * 16;
            if (kk + 32 < 384) {        // prefetch next K chunk -> global_prefetch_b8
                __builtin_prefetch(sx + 32, 0, 3);
                __builtin_prefetch(sg + 32, 0, 3);
            }
            unsigned short* dst = lA + r * TS + hf * 16;
#pragma unroll
            for (int t = 0; t < 4; ++t) {
                float4 a4 = reinterpret_cast<const float4*>(sx)[t];
                float4 g4 = reinterpret_cast<const float4*>(sg)[t];
                dst[4 * t + 0] = f2bf(a4.x * g4.x);
                dst[4 * t + 1] = f2bf(a4.y * g4.y);
                dst[4 * t + 2] = f2bf(a4.z * g4.z);
                dst[4 * t + 3] = f2bf(a4.w * g4.w);
            }
        }
        {   // stage B: pre-converted bf16 weights, 32B copy
            const unsigned short* sw = wqkv + (size_t)(n0 + r) * 384 + kk + hf * 16;
            *reinterpret_cast<uint4*>(lB + r * TS + hf * 16)     = *reinterpret_cast<const uint4*>(sw);
            *reinterpret_cast<uint4*>(lB + r * TS + hf * 16 + 8) = *reinterpret_cast<const uint4*>(sw + 8);
        }
        __syncthreads();
        bf16x16 a = load_a_frag(lA, 16 * wave, 0, TS, lane);
#pragma unroll
        for (int j = 0; j < 4; ++j) {
            bf16x16 b = load_b_frag_rows(lB, 16 * j, 0, TS, lane);
            acc[j] = wmma_bf16(a, b, acc[j]);
        }
        __syncthreads();
    }

    const int half = lane >> 4, nl = lane & 15;
#pragma unroll
    for (int j = 0; j < 4; ++j) {
#pragma unroll
        for (int r = 0; r < 8; ++r) {
            int m = m0 + 16 * wave + r + 8 * half;
            int n = n0 + 16 * j + nl;
            float v = acc[j][r] + qkv_b[n];
            if (n < 384) v *= SCALE_Q;                 // scale folded into stored q
            qkv_out[(size_t)m * 1152 + n] = f2bf(v);
        }
    }
}

// ---------------- Kernel 2: per-(b,h) attention: softmax(q k^T + bias + mask) v
__global__ __launch_bounds__(128) void attn_kernel(const unsigned short* __restrict__ qkv,
                                                   const float* __restrict__ mask,
                                                   const float* __restrict__ table,
                                                   const int*   __restrict__ rel_idx,
                                                   unsigned short* __restrict__ attn_out) {
    __shared__ unsigned short lq[64 * TS], lk[64 * TS], lv[64 * TS];
    __shared__ float S[64][65];
    __shared__ unsigned short P[64 * 72];   // 144B row stride: 16B-aligned chunks

    const int tid = threadIdx.x, lane = tid & 31, wave = tid >> 5;
    const int h = blockIdx.x, b = blockIdx.y;
    const size_t base = (size_t)b * 49 * 1152;

    // Stage q,k,v tiles (49x32) via Tensor Data Mover (one wave issues 3 DMAs).
    if (wave == 0) {
        tdm_load_qkv_tile((unsigned)(size_t)(void*)lq, qkv + base + h * 32);
        tdm_load_qkv_tile((unsigned)(size_t)(void*)lk, qkv + base + 384 + h * 32);
        tdm_load_qkv_tile((unsigned)(size_t)(void*)lv, qkv + base + 768 + h * 32);
    }
    // Zero the pad rows (49..63, cols 0..31) while DMAs are in flight.
    for (int i = tid; i < 15 * 4 * 3; i += 128) {
        int mat = i / 60, rem = i % 60;
        int rr = 49 + rem / 4, ch = rem % 4;
        unsigned short* bp = (mat == 0 ? lq : (mat == 1 ? lk : lv)) + rr * TS + ch * 8;
        uint4 z = {0, 0, 0, 0};
        *reinterpret_cast<uint4*>(bp) = z;
    }
    if (wave == 0) __builtin_amdgcn_s_wait_tensorcnt(0);
    __syncthreads();

    {   // S = q @ k^T (head_dim=32 -> single K step per tile)
        bf16x16 a = load_a_frag(lq, 16 * wave, 0, TS, lane);
        const int half = lane >> 4, nl = lane & 15;
#pragma unroll
        for (int j = 0; j < 4; ++j) {
            bf16x16 bf = load_b_frag_rows(lk, 16 * j, 0, TS, lane);
            f32x8 c = {};
            c = wmma_bf16(a, bf, c);
#pragma unroll
            for (int r = 0; r < 8; ++r)
                S[16 * wave + r + 8 * half][16 * j + nl] = c[r];
        }
    }
    __syncthreads();

    // softmax over 49 keys with relative-position bias gather + mask (row-per-thread)
    if (tid < 49) {
        const int i = tid;
        const float* mrow = mask + ((size_t)b * 49 + i) * 49;
        const int*   irow = rel_idx + i * 49;
        float mx = -1e30f;
        for (int j = 0; j < 49; ++j) {
            float s = S[i][j] + table[irow[j] * 12 + h] + mrow[j];
            S[i][j] = s;
            mx = fmaxf(mx, s);
        }
        float sum = 0.f;
        for (int j = 0; j < 49; ++j) {
            float e = __expf(S[i][j] - mx);
            S[i][j] = e;
            sum += e;
        }
        float inv = 1.f / sum;
        for (int j = 0; j < 49; ++j) P[i * 72 + j] = f2bf(S[i][j] * inv);
        for (int j = 49; j < 64; ++j) P[i * 72 + j] = 0;
    } else if (tid < 64) {
        for (int j = 0; j < 64; ++j) P[tid * 72 + j] = 0;
    }
    __syncthreads();

    {   // out = P @ v   (M=64 pad, K=64 pad, N=32)
        const int half = lane >> 4, nl = lane & 15;
#pragma unroll
        for (int jn = 0; jn < 2; ++jn) {
            f32x8 c = {};
#pragma unroll
            for (int ks = 0; ks < 2; ++ks) {
                bf16x16 a  = load_a_frag(P, 16 * wave, 32 * ks, 72, lane);
                bf16x16 bf = load_b_frag_cols(lv, 16 * jn, 32 * ks, TS, lane);
                c = wmma_bf16(a, bf, c);
            }
#pragma unroll
            for (int r = 0; r < 8; ++r) {
                int row = 16 * wave + r + 8 * half;
                if (row < 49) {
                    int d = 16 * jn + nl;
                    attn_out[((size_t)b * 49 + row) * 384 + h * 32 + d] = f2bf(c[r]);
                }
            }
        }
    }
}

// ---------------- Kernel 3: out = attn_out @ proj_w^T + proj_b (fp32 out)
__global__ __launch_bounds__(128) void proj_kernel(const unsigned short* __restrict__ A,
                                                   const unsigned short* __restrict__ wproj,
                                                   const float* __restrict__ proj_b,
                                                   float* __restrict__ out) {
    __shared__ unsigned short lA[64 * TS];
    __shared__ unsigned short lB[64 * TS];
    const int tid = threadIdx.x, lane = tid & 31, wave = tid >> 5;
    const int m0 = blockIdx.y * 64;
    const int n0 = blockIdx.x * 64;
    f32x8 acc[4] = {};

    for (int kk = 0; kk < 384; kk += 32) {
        const int r = tid >> 1, hf = tid & 1;
        {   // A already bf16: 32B copy (+ prefetch next chunk)
            const unsigned short* src = A + (size_t)(m0 + r) * 384 + kk + hf * 16;
            if (kk + 32 < 384) __builtin_prefetch(src + 32, 0, 3);
            *reinterpret_cast<uint4*>(lA + r * TS + hf * 16)     = *reinterpret_cast<const uint4*>(src);
            *reinterpret_cast<uint4*>(lA + r * TS + hf * 16 + 8) = *reinterpret_cast<const uint4*>(src + 8);
        }
        {   // B: pre-converted bf16 weights
            const unsigned short* sw = wproj + (size_t)(n0 + r) * 384 + kk + hf * 16;
            *reinterpret_cast<uint4*>(lB + r * TS + hf * 16)     = *reinterpret_cast<const uint4*>(sw);
            *reinterpret_cast<uint4*>(lB + r * TS + hf * 16 + 8) = *reinterpret_cast<const uint4*>(sw + 8);
        }
        __syncthreads();
        bf16x16 a = load_a_frag(lA, 16 * wave, 0, TS, lane);
#pragma unroll
        for (int j = 0; j < 4; ++j) {
            bf16x16 b = load_b_frag_rows(lB, 16 * j, 0, TS, lane);
            acc[j] = wmma_bf16(a, b, acc[j]);
        }
        __syncthreads();
    }

    const int half = lane >> 4, nl = lane & 15;
#pragma unroll
    for (int j = 0; j < 4; ++j) {
#pragma unroll
        for (int r = 0; r < 8; ++r) {
            int m = m0 + 16 * wave + r + 8 * half;
            int n = n0 + 16 * j + nl;
            out[(size_t)m * 384 + n] = acc[j][r] + proj_b[n];
        }
    }
}

extern "C" void kernel_launch(void* const* d_in, const int* in_sizes, int n_in,
                              void* d_out, int out_size, void* d_ws, size_t ws_size,
                              hipStream_t stream) {
    const float* x       = (const float*)d_in[0];
    const float* mask    = (const float*)d_in[1];
    const float* stagein = (const float*)d_in[2];
    const float* table   = (const float*)d_in[3];
    const int*   rel_idx = (const int*)  d_in[4];
    const float* qkv_w   = (const float*)d_in[5];
    const float* qkv_b   = (const float*)d_in[6];
    const float* proj_w  = (const float*)d_in[7];
    const float* proj_b  = (const float*)d_in[8];

    // workspace layout (all offsets 256B aligned)
    char* ws = (char*)d_ws;
    unsigned short* qkv_ws  = (unsigned short*)ws;                              // 231,211,008 B
    unsigned short* attn_ws = (unsigned short*)(ws + (size_t)M_TOT * 1152 * 2); //  77,070,336 B
    unsigned short* wqkv_bf = (unsigned short*)(ws + 308281344ull);             //     884,736 B
    unsigned short* wproj_bf= (unsigned short*)(ws + 309166080ull);             //     294,912 B
    float* out = (float*)d_out;

    dim3 blk(128);
    convert_weights<<<576, 256, 0, stream>>>(qkv_w, proj_w, wqkv_bf, wproj_bf);
    qkv_kernel <<<dim3(18, 1568), blk, 0, stream>>>(x, stagein, wqkv_bf, qkv_b, qkv_ws);
    attn_kernel<<<dim3(12, 2048), blk, 0, stream>>>(qkv_ws, mask, table, rel_idx, attn_ws);
    proj_kernel<<<dim3(6, 1568),  blk, 0, stream>>>(attn_ws, wproj_bf, proj_b, out);
}
